// BatchTopKTiedSAE_38457137168856
// MI455X (gfx1250) — matrix operand order repaired and checked by reading, not compiled, not run on hardware
//
#include <hip/hip_runtime.h>
#include <hip/hip_bf16.h>
#include <stdint.h>

// Problem constants (match the reference).
#define BATCH 4096
#define D_IN  2048
#define D_HID 16384
#define NBINS 2048          // histogram bins = float bits >> 20 (positive floats, monotonic)
#define NQ    (D_IN / 32)   // 64 K-steps for the encode GEMM

typedef __bf16 bf16;
typedef __bf16 v16bf __attribute__((ext_vector_type(16)));
typedef float  v8f   __attribute__((ext_vector_type(8)));

// ---------------------------------------------------------------------------
// CDNA5 async global->LDS copy (ASYNCcnt-tracked), per CDNA5 ISA.
// the LDS address comes from the VDST VGPR; low 32 bits of a flat shared pointer are the
// wave-relative LDS byte offset, so truncation gives the VDST value.
// ---------------------------------------------------------------------------
__device__ __forceinline__ unsigned lds_off(const void* p) {
  return (unsigned)(size_t)p;
}
__device__ __forceinline__ void async_load_b128(unsigned lds_byte_off,
                                                const void* gaddr) {
  asm volatile("global_load_async_to_lds_b128 %0, %1, off"
               :: "v"(lds_byte_off), "v"((unsigned long long)gaddr)
               : "memory");
}
__device__ __forceinline__ void wait_asynccnt0() {
  asm volatile("s_wait_asynccnt 0x0" ::: "memory");
}

// ---------------------------------------------------------------------------
// Pack x [BATCH][D_IN] f32 into WMMA-A fragment-ready bf16 layout:
// Xp[(Mtile*NQ + q)*32 + lane][16] = the 16 bf16 values lane `lane` feeds to
// v_wmma_f32_16x16x32_bf16 for A tile (Mtile, q).
// A 16x32 bf16 layout: lanes 0-15 hold M=lane, K {0..7,16..23};
//                      lanes 16-31 hold M=lane-16, K {8..15,24..31}.
// ---------------------------------------------------------------------------
__global__ __launch_bounds__(256)
void pack_x_kernel(const float* __restrict__ x, bf16* __restrict__ Xp) {
  int t = blockIdx.x * 256 + threadIdx.x;   // 0 .. 256*NQ*32-1
  int lane = t & 31;
  int q = (t >> 5) & (NQ - 1);
  int M = t >> 11;
  int row = lane & 15;
  int hi = (lane < 16) ? 0 : 8;
  const float* __restrict__ src = x + (size_t)(M * 16 + row) * D_IN + q * 32;
  bf16* __restrict__ dst = Xp + ((size_t)(M * NQ + q) * 32 + lane) * 16;
#pragma unroll
  for (int j = 0; j < 8; ++j) dst[j] = (bf16)src[hi + j];
#pragma unroll
  for (int j = 0; j < 8; ++j) dst[8 + j] = (bf16)src[16 + hi + j];
}

// ---------------------------------------------------------------------------
// Pack W [D_IN][D_HID] f32 into WMMA-B fragment-ready bf16 layout:
// Wp[(Ntile*NQ + q)*32 + lane][16].
// B 32x16 bf16 layout: lanes 0-15 hold N=lane, K {0..7,16..23}; lanes 16-31 K {8..15,24..31}.
// ---------------------------------------------------------------------------
__global__ __launch_bounds__(256)
void pack_w_kernel(const float* __restrict__ W, bf16* __restrict__ Wp) {
  int t = blockIdx.x * 256 + threadIdx.x;   // 0 .. 1024*NQ*32-1
  int lane = t & 31;
  int q = (t >> 5) & (NQ - 1);
  int N = t >> 11;
  int col = N * 16 + (lane & 15);
  int khi = (lane < 16) ? 0 : 8;
  const float* __restrict__ src = W + (size_t)(q * 32 + khi) * D_HID + col;
  bf16* __restrict__ dst = Wp + ((size_t)(N * NQ + q) * 32 + lane) * 16;
#pragma unroll
  for (int j = 0; j < 8; ++j) dst[j] = (bf16)src[(size_t)j * D_HID];
#pragma unroll
  for (int j = 0; j < 8; ++j) dst[8 + j] = (bf16)src[(size_t)(16 + j) * D_HID];
}

// ---------------------------------------------------------------------------
// W [D_IN][D_HID] fp32 -> WT [D_HID][D_IN] bf16 (64 MB -> L2-resident for decode)
// ---------------------------------------------------------------------------
__global__ __launch_bounds__(256)
void transpose_f32_to_bf16_kernel(const float* __restrict__ W, bf16* __restrict__ WT) {
  __shared__ float tile[32][33];
  int x  = blockIdx.x * 32 + threadIdx.x;   // d_hid
  int y0 = blockIdx.y * 32;                 // d_in
  for (int j = threadIdx.y; j < 32; j += 8)
    tile[j][threadIdx.x] = W[(size_t)(y0 + j) * D_HID + x];
  __syncthreads();
  int xo  = y0 + threadIdx.x;               // output col (d_in)
  int yo0 = blockIdx.x * 32;                // output row (d_hid)
  for (int j = threadIdx.y; j < 32; j += 8)
    WT[(size_t)(yo0 + j) * D_IN + xo] = (bf16)tile[threadIdx.x][j];
}

// ---------------------------------------------------------------------------
// Encode GEMM: F = relu(x @ W + b_enc).
// Block = 256 threads = 8 wave32 arranged 2(M) x 4(N); block tile 64M x 256N.
// Each wave: 32M x 64N strip = 2x4 accumulators, 8 wmma per K-step with
// 12 ds_load_b128 (2 A frags + 4 B frags). Operands are double-buffered in
// LDS and filled with global_load_async_to_lds_b128 one K-step ahead, so the
// s_wait_asynccnt 0 lands after a full wmma phase.
// ---------------------------------------------------------------------------
__global__ __launch_bounds__(256)
void sae_encode_wmma(const bf16* __restrict__ Xp, const bf16* __restrict__ Wp,
                     const float* __restrict__ b_enc, float* __restrict__ F) {
  __shared__ __align__(32) bf16 As[2 * 4 * 32 * 16];    // 2 bufs x 4 m-tiles  = 8KB
  __shared__ __align__(32) bf16 Bs[2 * 16 * 32 * 16];   // 2 bufs x 16 n-tiles = 32KB
  const int tid  = threadIdx.x;
  const int wave = tid >> 5;
  const int lane = tid & 31;
  const int mw = wave >> 2;                 // 0..1 (M wave, 2 m-tiles each)
  const int nw = wave & 3;                  // 0..3 (N wave, 4 n-tiles each)
  const int mtile0 = blockIdx.y * 4;
  const int ntile0 = blockIdx.x * 16;

  v8f acc[2][4] = {};

  // Per-thread staging chunk assignment (16B granularity).
  const int a_mt  = tid >> 6;               // A: 256 chunks (4 tiles x 64)
  const int a_sub = tid & 63;
  const unsigned as_base = lds_off(As);
  const unsigned bs_base = lds_off(Bs);

  // Issue one K-step of async fills into buffer `buf`.
  auto stage_async = [&](int q, int buf) {
    async_load_b128(as_base + (unsigned)(buf * 4096 + tid * 16),
                    Xp + ((size_t)(mtile0 + a_mt) * NQ + q) * (32 * 16) + a_sub * 8);
#pragma unroll
    for (int it = 0; it < 4; ++it) {
      int c = tid + it * 256;               // B: 1024 chunks (16 tiles x 64)
      int nt = c >> 6;
      int sub = c & 63;
      async_load_b128(bs_base + (unsigned)(buf * 16384 + c * 16),
                      Wp + ((size_t)(ntile0 + nt) * NQ + q) * (32 * 16) + sub * 8);
    }
  };

  stage_async(0, 0);
  wait_asynccnt0();
  __syncthreads();

  for (int q = 0; q < NQ; ++q) {
    const int cur = q & 1;
    if (q + 1 < NQ) stage_async(q + 1, cur ^ 1);  // fill next buffer, no wait yet

    // Fragments: contiguous 32B per lane -> 2 x ds_load_b128 each.
    v16bf a[2], b[4];
#pragma unroll
    for (int r = 0; r < 2; ++r)
      a[r] = ((const v16bf*)As)[cur * 128 + (mw * 2 + r) * 32 + lane];
#pragma unroll
    for (int i = 0; i < 4; ++i)
      b[i] = ((const v16bf*)Bs)[cur * 512 + (nw * 4 + i) * 32 + lane];

#pragma unroll
    for (int r = 0; r < 2; ++r)
#pragma unroll
      for (int i = 0; i < 4; ++i)
        acc[r][i] = __builtin_amdgcn_wmma_f32_16x16x32_bf16(
            false, a[r], false, b[i], (short)0, acc[r][i], false, false);

    wait_asynccnt0();   // next buffer filled (overlapped with the 8 wmmas)
    __syncthreads();
  }

  // C/D layout: VGPR e -> M = e (lanes 0-15) or e+8 (lanes 16-31), N = lane&15.
  const int ncol = lane & 15;
  const int mhi  = (lane < 16) ? 0 : 8;
#pragma unroll
  for (int r = 0; r < 2; ++r) {
    const int m0 = (mtile0 + mw * 2 + r) * 16;
#pragma unroll
    for (int i = 0; i < 4; ++i) {
      const int n0 = (ntile0 + nw * 4 + i) * 16;
      const float bias = b_enc[n0 + ncol];
#pragma unroll
      for (int e = 0; e < 8; ++e) {
        float v = acc[r][i][e] + bias;
        v = v > 0.f ? v : 0.f;  // fused ReLU
        F[(size_t)(m0 + mhi + e) * D_HID + n0 + ncol] = v;
      }
    }
  }
}

// ---------------------------------------------------------------------------
// Histogram of positive activations (bit-pattern monotonic for floats > 0).
// ---------------------------------------------------------------------------
__global__ __launch_bounds__(256)
void hist_kernel(const float* __restrict__ F, unsigned int* __restrict__ hist, long n4) {
  __shared__ unsigned int lh[NBINS];
  for (int i = threadIdx.x; i < NBINS; i += 256) lh[i] = 0;
  __syncthreads();
  const float4* __restrict__ F4 = (const float4*)F;
  long i = (long)blockIdx.x * blockDim.x + threadIdx.x;
  long stride = (long)gridDim.x * blockDim.x;
  for (; i < n4; i += stride) {
    float4 v = F4[i];
    if (v.x > 0.f) atomicAdd(&lh[__float_as_uint(v.x) >> 20], 1u);
    if (v.y > 0.f) atomicAdd(&lh[__float_as_uint(v.y) >> 20], 1u);
    if (v.z > 0.f) atomicAdd(&lh[__float_as_uint(v.z) >> 20], 1u);
    if (v.w > 0.f) atomicAdd(&lh[__float_as_uint(v.w) >> 20], 1u);
  }
  __syncthreads();
  for (int j = threadIdx.x; j < NBINS; j += 256)
    if (lh[j]) atomicAdd(&hist[j], lh[j]);
}

// Scan from the top bin down to find the keep-threshold for k*BATCH survivors.
__global__ void topk_threshold_kernel(const unsigned int* __restrict__ hist,
                                      const int* __restrict__ kptr,
                                      float* __restrict__ thresh) {
  if (threadIdx.x == 0 && blockIdx.x == 0) {
    long n_keep = (long)kptr[0] * BATCH;
    long cum = 0;
    int b = NBINS - 1;
    while (b > 0) {
      cum += hist[b];
      if (cum >= n_keep) break;
      --b;
    }
    *thresh = __uint_as_float((unsigned int)b << 20);
  }
}

// ---------------------------------------------------------------------------
// Sparse decode fused with top-k application. One block per batch row:
//  - thresholds the f row in place (producing f_topk in d_out),
//  - builds a deterministic (prefix-sum ordered) compact nonzero list,
//  - accumulates recon row (2048 f32 in LDS) from bf16 WT rows (L2-resident).
// Avg ~64 nonzeros/row -> ~1 GFLOP total vs 275 GFLOP dense.
// ---------------------------------------------------------------------------
__global__ __launch_bounds__(256)
void sae_decode_sparse(float* __restrict__ F, const bf16* __restrict__ WT,
                       const float* __restrict__ b_dec,
                       const float* __restrict__ thresh,
                       float* __restrict__ recon) {
  __shared__ float acc[D_IN];    // 8 KB recon accumulator
  __shared__ int   s_h[2048];    // worst case: whole chunk survives
  __shared__ float s_v[2048];
  __shared__ int   cnts[256];

  const int tid = threadIdx.x;
  const int b   = blockIdx.x;
  float* __restrict__ frow = F + (size_t)b * D_HID;

#pragma unroll
  for (int it = 0; it < D_IN / 256; ++it) acc[tid + it * 256] = 0.f;
  const float t = *thresh;

  for (int c0 = 0; c0 < D_HID; c0 += 2048) {
    // Each thread owns 8 consecutive columns -> h-ordered, deterministic list.
    const int h0 = c0 + tid * 8;
    float vv[8];
    int mycnt = 0;
#pragma unroll
    for (int j = 0; j < 8; ++j) {
      float v = frow[h0 + j];
      bool keep = (v >= t) && (v > 0.f);
      if (keep) {
        vv[j] = v;
        ++mycnt;
      } else {
        vv[j] = 0.f;
        if (v != 0.f) frow[h0 + j] = 0.f;  // zero dropped activation in f_topk
      }
    }
    cnts[tid] = mycnt;
    __syncthreads();
    int base = 0, total = 0;
    for (int i = 0; i < 256; ++i) {  // serial scan: 256 LDS reads, negligible
      int c = cnts[i];
      if (i < tid) base += c;
      total += c;
    }
    int slot = base;
#pragma unroll
    for (int j = 0; j < 8; ++j)
      if (vv[j] > 0.f) { s_h[slot] = h0 + j; s_v[slot] = vv[j]; ++slot; }
    __syncthreads();

    for (int e = 0; e < total; ++e) {
      const bf16* __restrict__ wrow = WT + (size_t)s_h[e] * D_IN;
      const float v = s_v[e];
#pragma unroll
      for (int it = 0; it < D_IN / 256; ++it) {
        int i = tid + it * 256;
        acc[i] += v * (float)wrow[i];
      }
    }
    __syncthreads();
  }

#pragma unroll
  for (int it = 0; it < D_IN / 256; ++it) {
    int i = tid + it * 256;
    recon[(size_t)b * D_IN + i] = acc[i] + b_dec[i];
  }
}

// ---------------------------------------------------------------------------
extern "C" void kernel_launch(void* const* d_in, const int* in_sizes, int n_in,
                              void* d_out, int out_size, void* d_ws, size_t ws_size,
                              hipStream_t stream) {
  (void)in_sizes; (void)n_in; (void)out_size; (void)ws_size;

  const float* x     = (const float*)d_in[0];
  const float* W     = (const float*)d_in[1];
  const float* b_enc = (const float*)d_in[2];
  const float* b_dec = (const float*)d_in[3];
  const int*   kptr  = (const int*)d_in[4];

  float* recon = (float*)d_out;                           // [BATCH, D_IN]
  float* F     = (float*)d_out + (size_t)BATCH * D_IN;    // [BATCH, D_HID] (f_topk)

  // Workspace layout.
  char* ws = (char*)d_ws;
  bf16* Xp = (bf16*)ws;                                           // 16 MB packed A
  bf16* Wp = (bf16*)(ws + (size_t)BATCH * D_IN * 2);              // 64 MB packed B
  bf16* WT = (bf16*)(ws + (size_t)BATCH * D_IN * 2
                        + (size_t)D_IN * D_HID * 2);              // 64 MB decode W^T
  unsigned int* hist = (unsigned int*)(ws + (size_t)BATCH * D_IN * 2
                                          + (size_t)D_IN * D_HID * 2 * 2);
  float* thresh = (float*)(hist + NBINS);

  // One-time operand packing (streams ~200 MB; L2-resident afterwards).
  pack_x_kernel<<<(BATCH / 16) * NQ * 32 / 256, 256, 0, stream>>>(x, Xp);
  pack_w_kernel<<<(D_HID / 16) * NQ * 32 / 256, 256, 0, stream>>>(W, Wp);
  transpose_f32_to_bf16_kernel<<<dim3(D_HID / 32, D_IN / 32), dim3(32, 8), 0, stream>>>(W, WT);

  // Dense bf16 WMMA encode (the only unavoidable 275 GFLOP of this model).
  sae_encode_wmma<<<dim3(D_HID / 256, BATCH / 64), 256, 0, stream>>>(Xp, Wp, b_enc, F);

  // Global batch top-k via bit-monotonic histogram + threshold.
  hipMemsetAsync(hist, 0, NBINS * sizeof(unsigned int), stream);
  hist_kernel<<<2048, 256, 0, stream>>>(F, hist, (long)BATCH * D_HID / 4);
  topk_threshold_kernel<<<1, 32, 0, stream>>>(hist, kptr, thresh);

  // Sparse decode (fused threshold apply): ~1 GFLOP, WT served from L2.
  sae_decode_sparse<<<BATCH, 256, 0, stream>>>(F, WT, b_dec, thresh, recon);
}
